// Head_29394756174034
// MI455X (gfx1250) — matrix operand order repaired and checked by reading, not compiled, not run on hardware
//
#include <hip/hip_runtime.h>

typedef __attribute__((ext_vector_type(16))) _Float16 v16h;
typedef __attribute__((ext_vector_type(8)))  float    v8f;

constexpr int Bb = 4, Tt = 4096, Cc = 1024, HS = 64;
constexpr int BT = Bb * Tt;

static __device__ __forceinline__ v8f wmma16(v16h a, v16h b, v8f c) {
  // D = A(16x32 f16) * B(32x16 f16) + C(16x16 f32)
  return __builtin_amdgcn_wmma_f32_16x16x32_f16(
      /*neg_a=*/false, a, /*neg_b=*/false, b,
      /*c_mod=*/(short)0, c, /*reuse_a=*/false, /*reuse_b=*/false);
}

static __device__ __forceinline__ v8f zero8() {
  v8f z = {0.f, 0.f, 0.f, 0.f, 0.f, 0.f, 0.f, 0.f};
  return z;
}

// ---------------------------------------------------------------------------
// Kernel 0: one-shot conversion of Wq/Wk/Wv (f32, [C][HS]) into f16 transposed
// [3][HS][C]. This makes every WMMA B-fragment in the projection kernel a
// single contiguous, 32B-aligned 16-half read (2x global_load_b128) instead of
// 16 strided global_load_b32. 384 KB total, L2-resident.
// ---------------------------------------------------------------------------
__global__ __launch_bounds__(256)
void wcvt_kernel(const float* __restrict__ Wq,
                 const float* __restrict__ Wk,
                 const float* __restrict__ Wv,
                 _Float16* __restrict__ wt) {
  const int idx = blockIdx.x * 256 + threadIdx.x;  // over HS*Cc
  const int m = blockIdx.y;                        // which matrix
  const float* W = (m == 0) ? Wq : (m == 1) ? Wk : Wv;
  const int n = idx / Cc;
  const int c = idx % Cc;
  wt[((size_t)m * HS + n) * Cc + c] = (_Float16)W[(size_t)c * HS + n];
}

// ---------------------------------------------------------------------------
// Kernel 1: fused QKV projection. One wave computes a 16-row x 64-col tile of
// Q, K and V, reading x exactly once (HBM roofline: 64 MB once). Outputs f16:
// Q,K row-major [BT,64], V transposed [B][64][T] so the attention kernel's
// B-fragments are contiguous.
// ---------------------------------------------------------------------------
__global__ __launch_bounds__(32)
void qkv_proj_kernel(const float* __restrict__ x,
                     const _Float16* __restrict__ wt,   // [3][HS][Cc] f16
                     _Float16* __restrict__ q16,
                     _Float16* __restrict__ k16,
                     _Float16* __restrict__ vt16) {
  const int lane = threadIdx.x & 31;
  const int g  = lane >> 4;   // half-wave group
  const int hl = lane & 15;
  const int row0 = blockIdx.x * 16;

  v8f accQ[4], accK[4], accV[4];
#pragma unroll
  for (int t = 0; t < 4; ++t) { accQ[t] = zero8(); accK[t] = zero8(); accV[t] = zero8(); }

  // A-fragment source row for this lane (both half-groups hold the same row,
  // different K-chunks).
  const float* xrow = x + (size_t)(row0 + hl) * Cc;
  const _Float16* wtq = wt;
  const _Float16* wtk = wt + (size_t)HS * Cc;
  const _Float16* wtv = wt + (size_t)2 * HS * Cc;

  for (int kk = 0; kk < Cc; kk += 32) {
    // A fragment: two contiguous, 32B-aligned 8-float chunks -> f16.
    const int b0 = kk + g * 16;
    const int b1 = kk + g * 16 + 8;
    v8f xa = *(const v8f*)(xrow + b0);
    v8f xb = *(const v8f*)(xrow + b1);
    if (kk + 32 < Cc) __builtin_prefetch(xrow + b0 + 32, 0, 3);
    v16h a;
#pragma unroll
    for (int i = 0; i < 8; ++i) { a[i] = (_Float16)xa[i]; a[8 + i] = (_Float16)xb[i]; }

    const int krow = kk + g * 16;  // B-fragment k base for this lane
#pragma unroll
    for (int t = 0; t < 4; ++t) {
      const int n = t * 16 + hl;   // B-fragment column for this lane
      const size_t woff = (size_t)n * Cc + krow;   // 32B-aligned in halves
      v16h bq = *(const v16h*)(wtq + woff);
      v16h bk = *(const v16h*)(wtk + woff);
      v16h bv = *(const v16h*)(wtv + woff);
      accQ[t] = wmma16(a, bq, accQ[t]);
      accK[t] = wmma16(a, bk, accK[t]);
      accV[t] = wmma16(a, bv, accV[t]);
    }
  }

  // C/D layout: VGPR r, lane l -> row m = r + 8*(l>>4), col n = (l&15)+16*t
  const int bidx  = row0 / Tt;
  const int tloc0 = row0 % Tt;
#pragma unroll
  for (int t = 0; t < 4; ++t) {
#pragma unroll
    for (int r = 0; r < 8; ++r) {
      const int m = r + g * 8;
      const int n = t * 16 + hl;
      q16[(size_t)(row0 + m) * HS + n] = (_Float16)accQ[t][r];
      k16[(size_t)(row0 + m) * HS + n] = (_Float16)accK[t][r];
      vt16[((size_t)bidx * HS + n) * Tt + (tloc0 + m)] = (_Float16)accV[t][r];
    }
  }
}

// ---------------------------------------------------------------------------
// Kernel 2: causal flash attention. One wave owns 16 query rows, streams key
// tiles of 32 with online softmax. Score WMMAs: Q(16x64) x K^T(64x32) as
// 2 k-steps x 2 key-halves. P re-shaped C-layout -> A-layout through LDS,
// then P(16x32) x V(32x64) as 4 N-tiles. K/V working set is L2-resident.
// ---------------------------------------------------------------------------
__global__ __launch_bounds__(32)
void attn_kernel(const _Float16* __restrict__ q16,
                 const _Float16* __restrict__ k16,
                 const _Float16* __restrict__ vt16,
                 float* __restrict__ out) {
  __shared__ _Float16 Plds[16 * 32];

  const int lane = threadIdx.x & 31;
  const int g  = lane >> 4;
  const int hl = lane & 15;
  const int b  = blockIdx.y;
  const int qb = blockIdx.x * 16;
  const size_t rowbase = (size_t)b * Tt;

  // Q A-fragments for both 32-wide k-steps over HS=64.
  v16h aQ[2];
  const _Float16* qrow = q16 + (rowbase + qb + hl) * HS;
#pragma unroll
  for (int s = 0; s < 2; ++s) {
    const int h0 = s * 32 + g * 16;
    const int h1 = s * 32 + g * 16 + 8;
#pragma unroll
    for (int i = 0; i < 8; ++i) {
      aQ[s][i]     = qrow[h0 + i];
      aQ[s][8 + i] = qrow[h1 + i];
    }
  }

  float rm[8], rs[8];
#pragma unroll
  for (int r = 0; r < 8; ++r) { rm[r] = -3.0e30f; rs[r] = 0.f; }
  v8f o[4];
#pragma unroll
  for (int t = 0; t < 4; ++t) o[t] = zero8();

  const float scale = 0.125f;           // 1/sqrt(64)
  const int nkt = (qb + 16 + 31) / 32;  // key tiles covering keys <= qb+15

  for (int kt = 0; kt < nkt; ++kt) {
    const int kb = kt * 32;

    // ---- scores: S0 = keys kb..kb+15, S1 = keys kb+16..kb+31
    v8f S0 = zero8(), S1 = zero8();
    const _Float16* krow0 = k16 + (rowbase + kb + hl) * HS;
    const _Float16* krow1 = k16 + (rowbase + kb + 16 + hl) * HS;
#pragma unroll
    for (int s = 0; s < 2; ++s) {
      const int h = s * 32 + g * 16;      // 32B-aligned contiguous run
      v16h bk0 = *(const v16h*)(krow0 + h);
      v16h bk1 = *(const v16h*)(krow1 + h);
      S0 = wmma16(aQ[s], bk0, S0);
      S1 = wmma16(aQ[s], bk1, S1);
    }

    // ---- causal mask + online softmax (per-row across 16-lane half-groups)
#pragma unroll
    for (int r = 0; r < 8; ++r) {
      const int mg = qb + r + g * 8;      // global query row for this element
      float s0 = S0[r] * scale;
      float s1 = S1[r] * scale;
      if (kb + hl > mg)      s0 = -3.0e30f;
      if (kb + 16 + hl > mg) s1 = -3.0e30f;

      float mx = fmaxf(s0, s1);
#pragma unroll
      for (int w = 1; w < 16; w <<= 1) mx = fmaxf(mx, __shfl_xor(mx, w, 32));
      const float mnew = fmaxf(rm[r], mx);
      const float corr = __expf(rm[r] - mnew);
      const float p0 = __expf(s0 - mnew);
      const float p1 = __expf(s1 - mnew);
      float sum = p0 + p1;
#pragma unroll
      for (int w = 1; w < 16; w <<= 1) sum += __shfl_xor(sum, w, 32);
      rs[r] = rs[r] * corr + sum;
      rm[r] = mnew;
#pragma unroll
      for (int t = 0; t < 4; ++t) o[t][r] *= corr;

      const int m = r + g * 8;
      Plds[m * 32 + hl]      = (_Float16)p0;
      Plds[m * 32 + 16 + hl] = (_Float16)p1;
    }
    __syncthreads();

    // ---- re-gather P as an A-fragment (16x32, K-dim = key)
    v16h aP;
#pragma unroll
    for (int i = 0; i < 8; ++i) {
      aP[i]     = Plds[hl * 32 + g * 16 + i];
      aP[8 + i] = Plds[hl * 32 + g * 16 + 8 + i];
    }
    __syncthreads();

    // ---- o += P * V  (V transposed: contiguous 16-half runs over key dim)
    const _Float16* vb = vt16 + (size_t)b * HS * Tt;
#pragma unroll
    for (int t = 0; t < 4; ++t) {
      const int n = t * 16 + hl;
      v16h bv = *(const v16h*)(vb + (size_t)n * Tt + kb + g * 16);
      o[t] = wmma16(aP, bv, o[t]);
    }
  }

  // ---- normalize and store fp32 output [B,T,HS]
#pragma unroll
  for (int r = 0; r < 8; ++r) {
    const int m = r + g * 8;
    const float inv = 1.0f / rs[r];
#pragma unroll
    for (int t = 0; t < 4; ++t) {
      out[(rowbase + qb + m) * HS + t * 16 + hl] = o[t][r] * inv;
    }
  }
}

extern "C" void kernel_launch(void* const* d_in, const int* in_sizes, int n_in,
                              void* d_out, int out_size, void* d_ws, size_t ws_size,
                              hipStream_t stream) {
  (void)in_sizes; (void)n_in; (void)out_size; (void)ws_size;
  const float* x  = (const float*)d_in[0];
  const float* Wq = (const float*)d_in[1];
  const float* Wk = (const float*)d_in[2];
  const float* Wv = (const float*)d_in[3];
  float* out = (float*)d_out;

  // Workspace layout (f16): Q[BT*64] | K[BT*64] | Vt[B*64*T] | Wt[3*64*1024]
  // = 6 MB + 384 KB.
  _Float16* q16  = (_Float16*)d_ws;
  _Float16* k16  = q16 + (size_t)BT * HS;
  _Float16* vt16 = k16 + (size_t)BT * HS;
  _Float16* wt   = vt16 + (size_t)Bb * HS * Tt;

  wcvt_kernel<<<dim3((HS * Cc) / 256, 3), dim3(256), 0, stream>>>(Wq, Wk, Wv, wt);
  qkv_proj_kernel<<<dim3(BT / 16), dim3(32), 0, stream>>>(x, wt, q16, k16, vt16);
  attn_kernel<<<dim3(Tt / 16, Bb), dim3(32), 0, stream>>>(q16, k16, vt16, out);
}
